// MultiHeadAttention_51410758533198
// MI455X (gfx1250) — matrix operand order repaired and checked by reading, not compiled, not run on hardware
//
#include <hip/hip_runtime.h>

// MHA: out = softmax_causal((xWq+bq)(xWk+bk)^T / sqrt(64)) (xWv+bv) Wo + bo
// B=4, S=2048, D=1024, H=16, DH=64.

#define Bc  4
#define Sc  2048
#define Dc  1024
#define Hc  16
#define DHc 64
#define GM  (Bc * Sc)   // 8192 rows

typedef __attribute__((ext_vector_type(16))) __bf16 v16bf;
typedef __attribute__((ext_vector_type(8)))  float  v8f;

union ABfrag { v16bf v; unsigned int u[8]; };

__device__ __forceinline__ unsigned short f2bf(float f) {
    unsigned int u = __builtin_bit_cast(unsigned int, f);
    u += 0x7FFFu + ((u >> 16) & 1u);   // round-to-nearest-even
    return (unsigned short)(u >> 16);
}

__device__ __forceinline__ float halfmax(float v) {
    v = fmaxf(v, __shfl_xor(v, 1));
    v = fmaxf(v, __shfl_xor(v, 2));
    v = fmaxf(v, __shfl_xor(v, 4));
    v = fmaxf(v, __shfl_xor(v, 8));
    return v;
}
__device__ __forceinline__ float halfsum(float v) {
    v += __shfl_xor(v, 1);
    v += __shfl_xor(v, 2);
    v += __shfl_xor(v, 4);
    v += __shfl_xor(v, 8);
    return v;
}

// ---------------------------------------------------------------------------
// Tiled bf16 WMMA GEMM: Out[M,N] = A[M,K] * W[K,N] + bias.
// MODE 0: bf16 out, head-major [B,H,S,DH].  MODE 2: f32 out, row-major [M,N].
// AF32: A is fp32 (converted while staging); else A is bf16 (ushort).
// Block = 256 thr (8 waves), tile 128x128, K step 32; wave does 32x64.
// ---------------------------------------------------------------------------
template<int MODE, bool AF32>
__global__ __launch_bounds__(256) void gemm128_bf16(
    const void* __restrict__ Aptr,
    const float* __restrict__ Wg,
    const float* __restrict__ bias,
    void* __restrict__ Out)
{
    __shared__ unsigned short As[128][32];   // [m][k]
    __shared__ unsigned short Bs[128][32];   // [n][k]  (W staged transposed)

    const int t    = threadIdx.x;
    const int lane = t & 31;
    const int wave = t >> 5;
    const int l15  = lane & 15;
    const int hiA  = (lane < 16) ? 0 : 8;    // accumulator hi-half row offset
    const int hiB  = (lane < 16) ? 0 : 16;   // B-frag hi-half K offset
    const int bm   = blockIdx.y * 128;
    const int bn   = blockIdx.x * 128;
    const int rm   = (wave & 3) * 32;        // wave's M offset in tile
    const int cn   = (wave >> 2) * 64;       // wave's N offset in tile
    const int Kd   = Dc, Nd = Dc;

    v8f acc[2][4];
#pragma unroll
    for (int mi = 0; mi < 2; ++mi)
#pragma unroll
        for (int ni = 0; ni < 4; ++ni)
            acc[mi][ni] = (v8f){0.f,0.f,0.f,0.f,0.f,0.f,0.f,0.f};

    for (int kk = 0; kk < Kd; kk += 32) {
        // ---- stage A tile (128x32) ----
        if (AF32) {
            const float* Ag = (const float*)Aptr;
#pragma unroll
            for (int i = 0; i < 4; ++i) {
                int idx = t + i * 256;             // 0..1023
                int row = idx >> 3, c4 = (idx & 7) * 4;
                const float4 f = *(const float4*)(Ag + (size_t)(bm + row) * Kd + kk + c4);
                As[row][c4 + 0] = f2bf(f.x);
                As[row][c4 + 1] = f2bf(f.y);
                As[row][c4 + 2] = f2bf(f.z);
                As[row][c4 + 3] = f2bf(f.w);
            }
        } else {
            const unsigned short* Ag = (const unsigned short*)Aptr;
#pragma unroll
            for (int i = 0; i < 2; ++i) {
                int idx = t + i * 256;             // 0..511
                int row = idx >> 2, c = (idx & 3) * 8;
                *(uint4*)&As[row][c] =
                    *(const uint4*)(Ag + (size_t)(bm + row) * Kd + kk + c);
            }
        }
        // ---- stage W tile (32x128) transposed to [n][k] ----
#pragma unroll
        for (int i = 0; i < 4; ++i) {
            int idx = t + i * 256;                 // 0..1023
            int kr = idx >> 5, c4 = (idx & 31) * 4;
            const float4 f = *(const float4*)(Wg + (size_t)(kk + kr) * Nd + bn + c4);
            Bs[c4 + 0][kr] = f2bf(f.x);
            Bs[c4 + 1][kr] = f2bf(f.y);
            Bs[c4 + 2][kr] = f2bf(f.z);
            Bs[c4 + 3][kr] = f2bf(f.w);
        }
        __syncthreads();

        // ---- A fragments (16x32 bf16, contiguous k = 16*half + s) ----
        ABfrag afr[2];
#pragma unroll
        for (int mi = 0; mi < 2; ++mi) {
            int row = rm + mi * 16 + l15;
#pragma unroll
            for (int v = 0; v < 8; ++v) {
                int k = 2 * v + hiB;
                afr[mi].u[v] = *(const unsigned int*)&As[row][k];
            }
        }
        // ---- B fragments (32x16 bf16) + WMMA ----
#pragma unroll
        for (int ni = 0; ni < 4; ++ni) {
            ABfrag bfr;
            int n = cn + ni * 16 + l15;
#pragma unroll
            for (int v = 0; v < 8; ++v)
                bfr.u[v] = *(const unsigned int*)&Bs[n][2 * v + hiB];
#pragma unroll
            for (int mi = 0; mi < 2; ++mi)
                acc[mi][ni] = __builtin_amdgcn_wmma_f32_16x16x32_bf16(
                    false, afr[mi].v, false, bfr.v, (short)0, acc[mi][ni],
                    false, false);
        }
        __syncthreads();
    }

    // ---- epilogue: bias + store ----
#pragma unroll
    for (int ni = 0; ni < 4; ++ni) {
        int n = bn + cn + ni * 16 + l15;
        float bv = bias[n];
#pragma unroll
        for (int mi = 0; mi < 2; ++mi) {
#pragma unroll
            for (int r = 0; r < 8; ++r) {
                int m = bm + rm + mi * 16 + r + hiA;
                float val = acc[mi][ni][r] + bv;
                if (MODE == 0) {
                    int b = m / Sc, s = m % Sc;
                    int h = n / DHc, d = n % DHc;
                    ((unsigned short*)Out)[(((size_t)(b * Hc + h)) * Sc + s) * DHc + d] =
                        f2bf(val);
                } else {
                    ((float*)Out)[(size_t)m * Nd + n] = val;
                }
            }
        }
    }
}

// ---------------------------------------------------------------------------
// Flash attention, bf16 WMMA, causal. grid = (S/128 q-tiles, B*H heads).
// Block = 256 thr (8 waves); each wave owns 16 q rows. Online softmax with
// shfl_xor half-wave row reductions; P re-shaped via per-wave LDS region.
// ---------------------------------------------------------------------------
__global__ __launch_bounds__(256) void flash_attn_bf16(
    const unsigned short* __restrict__ Qh,
    const unsigned short* __restrict__ Kh,
    const unsigned short* __restrict__ Vh,
    unsigned short* __restrict__ Xr)
{
    __shared__ unsigned short Qs[128][64];     // [q][d]
    __shared__ unsigned short Ks[128][64];     // [k][d]
    __shared__ unsigned short Vs[64][128];     // [d][k] (transposed)
    __shared__ unsigned short Ps[8][16][128];  // per-wave P tile [q][k]

    const int t    = threadIdx.x;
    const int lane = t & 31;
    const int w    = t >> 5;
    const int l15  = lane & 15;
    const int hi8  = (lane < 16) ? 0 : 8;
    const int hi16 = (lane < 16) ? 0 : 16;
    const int qt   = blockIdx.x;
    const int bh   = blockIdx.y;
    const int qbase = qt * 128;
    const int bIdx = bh / Hc, h = bh % Hc;
    const int qr   = w * 16;
    const size_t headoff = (size_t)bh * Sc * DHc;

    // ---- load Q tile ----
    {
        const unsigned short* Qg = Qh + headoff + (size_t)qbase * DHc;
#pragma unroll
        for (int i = 0; i < 4; ++i) {
            int idx = t + i * 256;
            int row = idx >> 3, c = (idx & 7) * 8;
            *(uint4*)&Qs[row][c] = *(const uint4*)(Qg + row * DHc + c);
        }
    }
    __syncthreads();

    // ---- Q fragments (constant over kt): two K-chunks of head dim ----
    ABfrag aq[2];
#pragma unroll
    for (int ci = 0; ci < 2; ++ci) {
        int row = qr + l15;
#pragma unroll
        for (int v = 0; v < 8; ++v) {
            int k = ci * 32 + 2 * v + hi16;
            aq[ci].u[v] = *(const unsigned int*)&Qs[row][k];
        }
    }

    float mrow[8], lrow[8];
    v8f Oacc[4];
#pragma unroll
    for (int r = 0; r < 8; ++r) { mrow[r] = -__builtin_inff(); lrow[r] = 0.f; }
#pragma unroll
    for (int dt = 0; dt < 4; ++dt)
        Oacc[dt] = (v8f){0.f,0.f,0.f,0.f,0.f,0.f,0.f,0.f};

    const unsigned short* Kg0 = Kh + headoff;
    const unsigned short* Vg0 = Vh + headoff;

    for (int kt = 0; kt <= qt; ++kt) {
        __syncthreads();  // previous tile fully consumed
        const unsigned short* Kg = Kg0 + (size_t)kt * 128 * DHc;
        const unsigned short* Vg = Vg0 + (size_t)kt * 128 * DHc;
#pragma unroll
        for (int i = 0; i < 4; ++i) {
            int idx = t + i * 256;
            int row = idx >> 3, c = (idx & 7) * 8;
            *(uint4*)&Ks[row][c] = *(const uint4*)(Kg + row * DHc + c);
            union { uint4 q; unsigned short s[8]; } tmp;
            tmp.q = *(const uint4*)(Vg + row * DHc + c);
#pragma unroll
            for (int j = 0; j < 8; ++j) Vs[c + j][row] = tmp.s[j];
        }
        if (kt < qt) {  // prefetch next K/V tile into cache (global_prefetch)
            __builtin_prefetch(Kg + 128 * DHc + t * 16, 0, 0);
            __builtin_prefetch(Vg + 128 * DHc + t * 16, 0, 0);
        }
        __syncthreads();

        // ---- scores: S(16x128) = Q(16x64) . K^T ----
        v8f sacc[8];
#pragma unroll
        for (int nt = 0; nt < 8; ++nt) {
            v8f c = (v8f){0.f,0.f,0.f,0.f,0.f,0.f,0.f,0.f};
#pragma unroll
            for (int ci = 0; ci < 2; ++ci) {
                ABfrag bk;
#pragma unroll
                for (int v = 0; v < 8; ++v)
                    bk.u[v] = *(const unsigned int*)
                        &Ks[nt * 16 + l15][ci * 32 + 2 * v + hi16];
                c = __builtin_amdgcn_wmma_f32_16x16x32_bf16(
                    false, aq[ci].v, false, bk.v, (short)0, c, false, false);
            }
            sacc[nt] = c;
        }

        // ---- causal mask + online softmax (per row r of this wave) ----
#pragma unroll
        for (int r = 0; r < 8; ++r) {
            const int rowg = qbase + qr + hi8 + r;
            float mx = -__builtin_inff();
#pragma unroll
            for (int nt = 0; nt < 8; ++nt) {
                int col = kt * 128 + nt * 16 + l15;
                float s = sacc[nt][r] * 0.125f;           // 1/sqrt(64)
                s = (col > rowg) ? -__builtin_inff() : s; // causal
                sacc[nt][r] = s;
                mx = fmaxf(mx, s);
            }
            mx = halfmax(mx);
            float mnew = fmaxf(mrow[r], mx);
            float corr = __expf(mrow[r] - mnew);
            float ps = 0.f;
#pragma unroll
            for (int nt = 0; nt < 8; ++nt) {
                float p = __expf(sacc[nt][r] - mnew);
                ps += p;
                Ps[w][hi8 + r][nt * 16 + l15] = f2bf(p);
            }
            ps = halfsum(ps);
            lrow[r] = lrow[r] * corr + ps;
            mrow[r] = mnew;
#pragma unroll
            for (int dt = 0; dt < 4; ++dt) Oacc[dt][r] *= corr;
        }

        // ---- O += P(16x128) . V(128x64) ---- (Ps region is wave-private)
#pragma unroll
        for (int kc = 0; kc < 4; ++kc) {
            ABfrag ap;
#pragma unroll
            for (int v = 0; v < 8; ++v) {
                int k = kc * 32 + 2 * v + hi16;
                ap.u[v] = *(const unsigned int*)&Ps[w][l15][k];
            }
#pragma unroll
            for (int dt = 0; dt < 4; ++dt) {
                ABfrag bv;
#pragma unroll
                for (int v = 0; v < 8; ++v)
                    bv.u[v] = *(const unsigned int*)
                        &Vs[dt * 16 + l15][kc * 32 + 2 * v + hi16];
                Oacc[dt] = __builtin_amdgcn_wmma_f32_16x16x32_bf16(
                    false, ap.v, false, bv.v, (short)0, Oacc[dt], false, false);
            }
        }
    }

    // ---- normalize and write x in [B,S,D] bf16 ----
#pragma unroll
    for (int dt = 0; dt < 4; ++dt) {
#pragma unroll
        for (int r = 0; r < 8; ++r) {
            int sg = qbase + qr + hi8 + r;
            float val = Oacc[dt][r] * (1.0f / lrow[r]);
            Xr[(size_t)(bIdx * Sc + sg) * Dc + h * DHc + dt * 16 + l15] = f2bf(val);
        }
    }
}

// ---------------------------------------------------------------------------
extern "C" void kernel_launch(void* const* d_in, const int* in_sizes, int n_in,
                              void* d_out, int out_size, void* d_ws, size_t ws_size,
                              hipStream_t stream)
{
    const float* query = (const float*)d_in[0];
    const float* key_  = (const float*)d_in[1];
    const float* value = (const float*)d_in[2];
    // d_in[3]: causal mask (tril) — causality applied analytically in-kernel.
    const float* Wq = (const float*)d_in[4];  const float* bq = (const float*)d_in[5];
    const float* Wk = (const float*)d_in[6];  const float* bk = (const float*)d_in[7];
    const float* Wv = (const float*)d_in[8];  const float* bv = (const float*)d_in[9];
    const float* Wo = (const float*)d_in[10]; const float* bo = (const float*)d_in[11];

    const size_t hsz = (size_t)Bc * Hc * Sc * DHc;   // 8M bf16 elems each
    unsigned short* Qh = (unsigned short*)d_ws;
    unsigned short* Kh = Qh + hsz;
    unsigned short* Vh = Kh + hsz;
    unsigned short* Xr = Vh + hsz;                   // [B,S,D] bf16

    dim3 gg(Dc / 128, GM / 128);                     // (8, 64)
    gemm128_bf16<0, true><<<gg, 256, 0, stream>>>(query, Wq, bq, Qh);
    gemm128_bf16<0, true><<<gg, 256, 0, stream>>>(key_, Wk, bk, Kh);
    gemm128_bf16<0, true><<<gg, 256, 0, stream>>>(value, Wv, bv, Vh);

    flash_attn_bf16<<<dim3(Sc / 128, Bc * Hc), 256, 0, stream>>>(Qh, Kh, Vh, Xr);

    gemm128_bf16<2, false><<<gg, 256, 0, stream>>>(Xr, Wo, bo, d_out);
}
